// DiffusionDocREModel_41128606826818
// MI455X (gfx1250) — matrix operand order repaired and checked by reading, not compiled, not run on hardware
//
#include <hip/hip_runtime.h>
#include <hip/hip_bf16.h>

// ---------------------------------------------------------------------------
// Problem constants (from the reference)
// ---------------------------------------------------------------------------
#define BS  4
#define LSEQ 1024
#define NH  12
#define DIM 768
#define NE  42
#define NM  96
#define NP  (NE * NE)      // 1764
#define OFF 1

typedef __attribute__((ext_vector_type(16))) _Float16 v16h;
typedef __attribute__((ext_vector_type(8)))  _Float16 v8h;
typedef __attribute__((ext_vector_type(8)))  float    v8f;

// ---------------------------------------------------------------------------
// K1: logsumexp-pool mention embeddings into entity embeddings
// grid (NE, BS), block 256.  e_emb[doc][e][d]
// ---------------------------------------------------------------------------
__global__ void entity_emb_kernel(const float* __restrict__ seq,
                                  const int* __restrict__ pos,
                                  const int* __restrict__ ent,
                                  float* __restrict__ e_emb) {
    const int e   = blockIdx.x;
    const int doc = blockIdx.y;
    __shared__ int s_pos[NM];
    __shared__ int s_ent[NM];
    if (threadIdx.x < NM) {
        s_pos[threadIdx.x] = pos[doc * NM + threadIdx.x] + OFF;
        s_ent[threadIdx.x] = ent[doc * NM + threadIdx.x];
    }
    __syncthreads();

    int cnt = 0;
    for (int m = 0; m < NM; ++m) cnt += (s_ent[m] == e);

    const float* sdoc = seq + (size_t)doc * LSEQ * DIM;
    float* out = e_emb + ((size_t)doc * NE + e) * DIM;

    #pragma unroll
    for (int i = 0; i < 3; ++i) {
        const int d = threadIdx.x + i * 256;
        float val = 0.0f;
        if (cnt > 0) {
            float mx = -3.402823466e+38f;
            for (int m = 0; m < NM; ++m)
                if (s_ent[m] == e) mx = fmaxf(mx, sdoc[(size_t)s_pos[m] * DIM + d]);
            float s = 0.0f;
            for (int m = 0; m < NM; ++m)
                if (s_ent[m] == e) s += expf(sdoc[(size_t)s_pos[m] * DIM + d] - mx);
            val = logf(fmaxf(s, 1e-30f)) + mx;
        }
        out[d] = val;
    }
}

// ---------------------------------------------------------------------------
// K2: mean-pool gathered attention rows into entity attention
// grid (NE*NH, BS), block 256 (4 L-values per thread). e_att[doc][e][h][l]
// ---------------------------------------------------------------------------
__global__ void entity_att_kernel(const float* __restrict__ att,
                                  const int* __restrict__ pos,
                                  const int* __restrict__ ent,
                                  float* __restrict__ e_att) {
    const int e   = blockIdx.x / NH;
    const int h   = blockIdx.x % NH;
    const int doc = blockIdx.y;
    __shared__ int s_pos[NM];
    __shared__ int s_ent[NM];
    if (threadIdx.x < NM) {
        s_pos[threadIdx.x] = pos[doc * NM + threadIdx.x] + OFF;
        s_ent[threadIdx.x] = ent[doc * NM + threadIdx.x];
    }
    __syncthreads();

    int cnt = 0;
    for (int m = 0; m < NM; ++m) cnt += (s_ent[m] == e);

    float acc[4] = {0.f, 0.f, 0.f, 0.f};
    for (int m = 0; m < NM; ++m) {
        if (s_ent[m] != e) continue;
        const float* row = att + (((size_t)doc * NH + h) * LSEQ + s_pos[m]) * LSEQ;
        #pragma unroll
        for (int i = 0; i < 4; ++i) acc[i] += row[threadIdx.x + i * 256];
    }
    const float scale = 1.0f / fmaxf((float)cnt, 1.0f);
    float* out = e_att + (((size_t)doc * NE + e) * NH + h) * LSEQ;
    #pragma unroll
    for (int i = 0; i < 4; ++i) out[threadIdx.x + i * 256] = acc[i] * scale;
}

// ---------------------------------------------------------------------------
// K3: seq [L,D] f32 -> seqT [D,L] f16 (B-operand staging for WMMA)
// grid (L, BS), block 256, 3 d per thread.
// ---------------------------------------------------------------------------
__global__ void seq_transpose_kernel(const float* __restrict__ seq,
                                     _Float16* __restrict__ seqT) {
    const int l   = blockIdx.x;
    const int doc = blockIdx.y;
    const float* src = seq + ((size_t)doc * LSEQ + l) * DIM;
    #pragma unroll
    for (int i = 0; i < 3; ++i) {
        const int d = threadIdx.x + i * 256;
        seqT[((size_t)doc * DIM + d) * LSEQ + l] = (_Float16)src[d];
    }
}

// ---------------------------------------------------------------------------
// K4: fused pair + normalize + f16 convert.
//   raw[l]   = (1/H) * sum_h e_att[a,h,l]*e_att[b,h,l]
//   pairH[l] = (f16)( raw[l] / (sum_l raw + 1e-5) )
// One block owns a full (doc, a, b) row of L=1024, so the row sum is
// available in-block via LDS reduction -> no f32 round-trip to HBM.
// grid (NE*NE, BS), block 256 (4 l per thread).
// ---------------------------------------------------------------------------
__global__ void pair_norm_kernel(const float* __restrict__ e_att,
                                 _Float16* __restrict__ pairH) {
    const int ab  = blockIdx.x;
    const int a   = ab / NE;
    const int b   = ab % NE;
    const int doc = blockIdx.y;
    const float* ea = e_att + (size_t)doc * NE * NH * LSEQ;
    _Float16* prow = pairH + ((size_t)doc * NP + ab) * LSEQ;

    float acc[4];
    float tot = 0.0f;
    #pragma unroll
    for (int i = 0; i < 4; ++i) {
        const int l = threadIdx.x + i * 256;
        float s = 0.0f;
        #pragma unroll
        for (int h = 0; h < NH; ++h)
            s += ea[((size_t)a * NH + h) * LSEQ + l] * ea[((size_t)b * NH + h) * LSEQ + l];
        s *= (1.0f / (float)NH);
        acc[i] = s;
        tot += s;
    }
    __shared__ float red[256];
    red[threadIdx.x] = tot;
    __syncthreads();
    for (int s = 128; s > 0; s >>= 1) {
        if (threadIdx.x < s) red[threadIdx.x] += red[threadIdx.x + s];
        __syncthreads();
    }
    const float inv = 1.0f / (red[0] + 1e-5f);
    #pragma unroll
    for (int i = 0; i < 4; ++i)
        prow[threadIdx.x + i * 256] = (_Float16)(acc[i] * inv);
}

// ---------------------------------------------------------------------------
// K5: feature_map GEMM: fm[ab,d] = sum_l pairH[ab,l] * seqT[d,l]
// Per doc: [1764 x 1024] x [1024 x 768] in f16 WMMA, f32 accumulate.
// grid (111, 6, BS), block 128 = 4 waves; wave w handles N-tiles
// blockIdx.y*8 + 2w and +1 with ONE shared A fragment (2 accumulators ->
// half the A traffic, two independent WMMA chains hide the RAW hazard).
//
// A fragment layout (16-bit A 16x32): lane r (0-15) rows; lanes 0-15 hold
// K 0-15, lanes 16-31 hold K 16-31 of the same rows (same k order as B)
//   -> one contiguous 32B (v16h) load at K0 + half*16.
// B fragment (32x16): lane holds column n = lane&15; K = (lane>>4)*16 + i
//   -> one contiguous 32B (v16h) load.
// D (f32 16x16): element j is row (lane>>4)*8 + j, column lane&15.
// ---------------------------------------------------------------------------
__global__ void gemm_feature_kernel(const _Float16* __restrict__ pairH,
                                    const _Float16* __restrict__ seqT,
                                    float* __restrict__ fm) {
    const int doc   = blockIdx.z;
    const int wave  = threadIdx.x >> 5;
    const int lane  = threadIdx.x & 31;
    const int half  = lane >> 4;
    const int r16   = lane & 15;

    const int tileM = blockIdx.x;                       // 0..110
    const int tileN0 = blockIdx.y * 8 + wave * 2;       // 0..46 (even)
    const int m  = tileM * 16 + r16;
    const int n0 = tileN0 * 16 + r16;
    const int n1 = n0 + 16;

    const _Float16* __restrict__ rowPtr =
        pairH + ((size_t)doc * NP + (m < NP ? m : 0)) * LSEQ;
    const _Float16* __restrict__ colPtr0 =
        seqT + ((size_t)doc * DIM + n0) * LSEQ;
    const _Float16* __restrict__ colPtr1 =
        seqT + ((size_t)doc * DIM + n1) * LSEQ;

    v8f c0 = {};
    v8f c1 = {};
    #pragma unroll 4
    for (int kt = 0; kt < LSEQ / 32; ++kt) {
        const int K0 = kt * 32;
        if (kt + 1 < LSEQ / 32) {
            __builtin_prefetch(rowPtr + K0 + 32, 0, 0);
            __builtin_prefetch(colPtr0 + K0 + 32, 0, 0);
            __builtin_prefetch(colPtr1 + K0 + 32, 0, 0);
        }
        // ---- A fragment: contiguous 32B load, K = half*16 + i (matches B) ----
        const v16h a = *reinterpret_cast<const v16h*>(rowPtr + K0 + half * 16);
        // ---- B fragments: contiguous 32B loads ----
        const v16h b0 = *reinterpret_cast<const v16h*>(colPtr0 + K0 + half * 16);
        const v16h b1 = *reinterpret_cast<const v16h*>(colPtr1 + K0 + half * 16);
        // ---- two independent accumulate chains ----
        c0 = __builtin_amdgcn_wmma_f32_16x16x32_f16(
                false, a, false, b0, (short)0, c0, false, false);
        c1 = __builtin_amdgcn_wmma_f32_16x16x32_f16(
                false, a, false, b1, (short)0, c1, false, false);
    }

    // ---- store two 16x16 f32 tiles ----
    const int mbase = tileM * 16 + half * 8;
    float* out = fm + (size_t)doc * NP * DIM;
    #pragma unroll
    for (int j = 0; j < 8; ++j) {
        const int mm = mbase + j;
        if (mm < NP) {
            out[(size_t)mm * DIM + n0] = c0[j];
            out[(size_t)mm * DIM + n1] = c1[j];
        }
    }
}

// ---------------------------------------------------------------------------
// K6: hss / tss gathers. grid (NP, BS), block 256, 3 d per thread.
// ---------------------------------------------------------------------------
__global__ void gather_ht_kernel(const float* __restrict__ e_emb,
                                 const int* __restrict__ hts,
                                 float* __restrict__ hss,
                                 float* __restrict__ tss) {
    const int p   = blockIdx.x;
    const int doc = blockIdx.y;
    const int h   = hts[((size_t)doc * NP + p) * 2 + 0];
    const int t   = hts[((size_t)doc * NP + p) * 2 + 1];
    const float* eh = e_emb + ((size_t)doc * NE + h) * DIM;
    const float* et = e_emb + ((size_t)doc * NE + t) * DIM;
    float* oh = hss + ((size_t)doc * NP + p) * DIM;
    float* ot = tss + ((size_t)doc * NP + p) * DIM;
    #pragma unroll
    for (int i = 0; i < 3; ++i) {
        const int d = threadIdx.x + i * 256;
        oh[d] = eh[d];
        ot[d] = et[d];
    }
}

// ---------------------------------------------------------------------------
// Launcher
// ---------------------------------------------------------------------------
static inline size_t align256(size_t x) { return (x + 255) & ~(size_t)255; }

extern "C" void kernel_launch(void* const* d_in, const int* in_sizes, int n_in,
                              void* d_out, int out_size, void* d_ws, size_t ws_size,
                              hipStream_t stream) {
    (void)in_sizes; (void)n_in; (void)out_size; (void)ws_size;

    const float* seq = (const float*)d_in[0];   // [BS, L, D]
    const float* att = (const float*)d_in[1];   // [BS, H, L, L]
    const int*   pos = (const int*)d_in[2];     // [BS, M]
    const int*   ent = (const int*)d_in[3];     // [BS, M]
    const int*   hts = (const int*)d_in[4];     // [BS, P, 2]

    // workspace layout (~30 MB)
    char* ws = (char*)d_ws;
    size_t off = 0;
    float* e_att = (float*)(ws + off);
    off = align256(off + (size_t)BS * NE * NH * LSEQ * sizeof(float));
    float* e_emb = (float*)(ws + off);
    off = align256(off + (size_t)BS * NE * DIM * sizeof(float));
    _Float16* pairH = (_Float16*)(ws + off);
    off = align256(off + (size_t)BS * NP * LSEQ * sizeof(_Float16));
    _Float16* seqT = (_Float16*)(ws + off);

    // output layout: hss | tss | feature_map (all f32)
    const size_t HS = (size_t)BS * NP * DIM;
    float* hss = (float*)d_out;
    float* tss = hss + HS;
    float* fm  = hss + 2 * HS;

    entity_emb_kernel<<<dim3(NE, BS), 256, 0, stream>>>(seq, pos, ent, e_emb);
    entity_att_kernel<<<dim3(NE * NH, BS), 256, 0, stream>>>(att, pos, ent, e_att);
    seq_transpose_kernel<<<dim3(LSEQ, BS), 256, 0, stream>>>(seq, seqT);
    pair_norm_kernel<<<dim3(NP, BS), 256, 0, stream>>>(e_att, pairH);
    gemm_feature_kernel<<<dim3((NP + 15) / 16, 6, BS), 128, 0, stream>>>(
        pairH, seqT, fm);
    gather_ht_kernel<<<dim3(NP, BS), 256, 0, stream>>>(e_emb, hts, hss, tss);
}
